// FeatureExtractionModel_sep_83846351552726
// MI455X (gfx1250) — compile-verified, matrix operand order checked by reading, not run on hardware
//
#include <hip/hip_runtime.h>
#include <hip/hip_bf16.h>
#include <stdint.h>

// ---------------------------------------------------------------------------
// Model dims
// ---------------------------------------------------------------------------
#define BB      4
#define NP      2048
#define NR      10
#define EMB     64
#define HEADS   16
#define NLAYERS 4
#define PROWS   (BB * NP)          // 8192
#define RROWS   (BB * NP * NR)     // 81920
#define PEP_F   (27 + 1280)        // 1307
#define REC_F   (42 + 1280)        // 1322
#define EDGE_F  32
#define LN_EPS  1e-5f

typedef __attribute__((ext_vector_type(16))) __bf16 v16bf;
typedef __attribute__((ext_vector_type(8)))  float  v8f;

// ---------------------------------------------------------------------------
// Repack weight W[64 x K] (row-major, f32) into WMMA B-fragment order (bf16):
// layout [K/32][t=0..3][lane=0..31][j=0..15]
//   n  = t*16 + (lane & 15)
//   kb = (lane >> 4) * 8
//   k  = ks*32 + (j < 8 ? kb + j : kb + 8 + j)
// ---------------------------------------------------------------------------
__global__ void repack_w_kernel(const float* __restrict__ W, __bf16* __restrict__ out, int K) {
    int idx = blockIdx.x * blockDim.x + threadIdx.x;
    int total = K * 64;                   // (K/32) * 4 * 32 * 16
    if (idx >= total) return;
    int j    = idx & 15;
    int lane = (idx >> 4) & 31;
    int t    = (idx >> 9) & 3;
    int ks   = idx >> 11;
    int n    = t * 16 + (lane & 15);
    int kb   = (lane >> 4) * 8;
    int k    = ks * 32 + ((j < 8) ? (kb + j) : (kb + 8 + j));
    out[idx] = (__bf16)W[(size_t)n * K + k];
}

// ---------------------------------------------------------------------------
// Generic GEMM: C[M x 64] = A[M x K] @ W^T  (+ bias), bf16 WMMA, f32 accum.
// A element type T (float or __bf16), row stride lda, K % 32 == 0.
// One wave -> 16 rows x 64 cols (4 accumulators). blockDim = 256 (8 waves),
// grid = M / 128.
// ---------------------------------------------------------------------------
template <typename T>
__global__ void gemm_n64_kernel(const T* __restrict__ A, long long lda,
                                const __bf16* __restrict__ Wfrag,
                                const float* __restrict__ bias,
                                float* __restrict__ C, int K) {
    const int lane   = threadIdx.x & 31;
    const int wave   = threadIdx.x >> 5;
    const int tile_m = (blockIdx.x * (blockDim.x >> 5) + wave) * 16;
    const int row    = lane & 15;
    const int kb     = (lane >> 4) * 8;

    v8f acc0 = {}, acc1 = {}, acc2 = {}, acc3 = {};

    const T* ap = A + (size_t)(tile_m + row) * (size_t)lda + kb;
    const int nsteps = K >> 5;

    for (int ks = 0; ks < nsteps; ++ks) {
        v16bf af;
#pragma unroll
        for (int j = 0; j < 8; ++j)  af[j] = (__bf16)(float)ap[j];
#pragma unroll
        for (int j = 8; j < 16; ++j) af[j] = (__bf16)(float)ap[j + 8];

        const v16bf* wf = ((const v16bf*)Wfrag) + (size_t)ks * 128 + lane;
        acc0 = __builtin_amdgcn_wmma_f32_16x16x32_bf16(false, af, false, wf[0],  (short)0, acc0, false, false);
        acc1 = __builtin_amdgcn_wmma_f32_16x16x32_bf16(false, af, false, wf[32], (short)0, acc1, false, false);
        acc2 = __builtin_amdgcn_wmma_f32_16x16x32_bf16(false, af, false, wf[64], (short)0, acc2, false, false);
        acc3 = __builtin_amdgcn_wmma_f32_16x16x32_bf16(false, af, false, wf[96], (short)0, acc3, false, false);
        ap += 32;
    }

    // D layout: lanes 0-15 -> (M = v,     N = lane);
    //           lanes16-31 -> (M = 8 + v, N = lane - 16)
    const int nl = lane & 15;
    const int mh = (lane >> 4) * 8;
    const float b0 = bias ? bias[nl]      : 0.0f;
    const float b1 = bias ? bias[16 + nl] : 0.0f;
    const float b2 = bias ? bias[32 + nl] : 0.0f;
    const float b3 = bias ? bias[48 + nl] : 0.0f;
#pragma unroll
    for (int v = 0; v < 8; ++v) {
        float* cr = C + (size_t)(tile_m + mh + v) * 64;
        cr[nl]      = acc0[v] + b0;
        cr[16 + nl] = acc1[v] + b1;
        cr[32 + nl] = acc2[v] + b2;
        cr[48 + nl] = acc3[v] + b3;
    }
}

// ---------------------------------------------------------------------------
// rec embedding (85 -> 64, LN) + edge embedding (32 -> 64), concatenated,
// stored as bf16 [RROWS x 128] (feeds the key/val WMMA GEMMs).
// grid = RROWS, block = 64.
// ---------------------------------------------------------------------------
__global__ void embed_rec_edge_kernel(const float* __restrict__ rec_feat,
                                      const float* __restrict__ rec_esm,
                                      const float* __restrict__ edge_feat,
                                      const float* __restrict__ W_rec_emb,
                                      const float* __restrict__ W_edge_emb,
                                      const float* __restrict__ g,
                                      const float* __restrict__ bt,
                                      __bf16* __restrict__ rec_edge) {
    const int rowid = blockIdx.x;
    const int d = threadIdx.x;
    const float* rf = rec_feat + (size_t)rowid * REC_F;
    const float* re = rec_esm + (size_t)rowid * 64;
    const float* wr = W_rec_emb + d * 85;
    float acc = 0.f;
    for (int k = 0; k < 21; ++k) acc += rf[k] * wr[k];
    for (int k = 0; k < 64; ++k) acc += re[k] * wr[21 + k];
    __shared__ float s[64];
    s[d] = acc;
    __syncthreads();
    float mu = 0.f;
    for (int k = 0; k < 64; ++k) mu += s[k];
    mu *= 0.015625f;
    float var = 0.f;
    for (int k = 0; k < 64; ++k) { float t = s[k] - mu; var += t * t; }
    var *= 0.015625f;
    float y = (acc - mu) * rsqrtf(var + LN_EPS) * g[d] + bt[d];

    const float* ef = edge_feat + (size_t)rowid * EDGE_F;
    const float* we = W_edge_emb + d * EDGE_F;
    float e = 0.f;
    for (int k = 0; k < EDGE_F; ++k) e += ef[k] * we[k];

    __bf16* out = rec_edge + (size_t)rowid * 128;
    out[d]      = (__bf16)y;
    out[64 + d] = (__bf16)e;
}

// ---------------------------------------------------------------------------
// pep embedding (91 -> 64, LN) -> f32 [PROWS x 64]. grid = PROWS, block = 64.
// ---------------------------------------------------------------------------
__global__ void embed_pep_kernel(const float* __restrict__ pep_feat,
                                 const float* __restrict__ pep_esm,
                                 const float* __restrict__ W_pep_emb,
                                 const float* __restrict__ g,
                                 const float* __restrict__ bt,
                                 float* __restrict__ pep) {
    const int rowid = blockIdx.x;
    const int d = threadIdx.x;
    const float* pf = pep_feat + (size_t)rowid * PEP_F;
    const float* pe = pep_esm + (size_t)rowid * 64;
    const float* w = W_pep_emb + d * 91;
    float acc = 0.f;
    for (int k = 0; k < 27; ++k) acc += pf[k] * w[k];
    for (int k = 0; k < 64; ++k) acc += pe[k] * w[27 + k];
    __shared__ float s[64];
    s[d] = acc;
    __syncthreads();
    float mu = 0.f;
    for (int k = 0; k < 64; ++k) mu += s[k];
    mu *= 0.015625f;
    float var = 0.f;
    for (int k = 0; k < 64; ++k) { float t = s[k] - mu; var += t * t; }
    var *= 0.015625f;
    pep[(size_t)rowid * 64 + d] = (acc - mu) * rsqrtf(var + LN_EPS) * g[d] + bt[d];
}

// ---------------------------------------------------------------------------
// Attention over NR=10 neighbors: pep = q + softmax(q . key / 8) @ val.
// grid = PROWS, block = 64.
// ---------------------------------------------------------------------------
__global__ void attention_kernel(const float* __restrict__ q,
                                 const float* __restrict__ key,
                                 const float* __restrict__ val,
                                 float* __restrict__ pep) {
    const int rowid = blockIdx.x;
    const int d = threadIdx.x;
    __shared__ float s[64];
    __shared__ float lg[NR];
    const float qd = q[(size_t)rowid * 64 + d];
    for (int r = 0; r < NR; ++r) {
        s[d] = qd * key[((size_t)rowid * NR + r) * 64 + d];
        __syncthreads();
        for (int off = 32; off > 0; off >>= 1) {
            if (d < off) s[d] += s[d + off];
            __syncthreads();
        }
        if (d == 0) lg[r] = s[0] * 0.125f;   // 1/sqrt(64)
        __syncthreads();
    }
    float mx = lg[0];
    for (int r = 1; r < NR; ++r) mx = fmaxf(mx, lg[r]);
    float w[NR];
    float se = 0.f;
    for (int r = 0; r < NR; ++r) { w[r] = __expf(lg[r] - mx); se += w[r]; }
    const float inv = 1.f / se;
    float acc = qd;
    for (int r = 0; r < NR; ++r)
        acc += w[r] * inv * val[((size_t)rowid * NR + r) * 64 + d];
    pep[(size_t)rowid * 64 + d] = acc;
}

// ---------------------------------------------------------------------------
// Pooling head: d = LN(pep); h = tanh(d W0^T); h = tanh(h W1^T); logits = h W2^T.
// grid = PROWS, block = 64. Writes dmat [PROWS x 64] and logits [PROWS x 16].
// ---------------------------------------------------------------------------
__global__ void pool_mlp_kernel(const float* __restrict__ pep,
                                const float* __restrict__ g,
                                const float* __restrict__ bt,
                                const float* __restrict__ W0,
                                const float* __restrict__ W1,
                                const float* __restrict__ W2,
                                float* __restrict__ dmat,
                                float* __restrict__ logits) {
    const int rowid = blockIdx.x;
    const int d = threadIdx.x;
    __shared__ float sd[64];
    __shared__ float h0[64];
    __shared__ float h1[32];
    const float x = pep[(size_t)rowid * 64 + d];
    sd[d] = x;
    __syncthreads();
    float mu = 0.f;
    for (int k = 0; k < 64; ++k) mu += sd[k];
    mu *= 0.015625f;
    float var = 0.f;
    for (int k = 0; k < 64; ++k) { float t = sd[k] - mu; var += t * t; }
    var *= 0.015625f;
    const float y = (x - mu) * rsqrtf(var + LN_EPS) * g[d] + bt[d];
    __syncthreads();
    sd[d] = y;
    __syncthreads();
    dmat[(size_t)rowid * 64 + d] = y;
    float a0 = 0.f;
    const float* w0 = W0 + d * 64;
    for (int k = 0; k < 64; ++k) a0 += sd[k] * w0[k];
    h0[d] = tanhf(a0);
    __syncthreads();
    if (d < 32) {
        float a1 = 0.f;
        const float* w1 = W1 + d * 64;
        for (int k = 0; k < 64; ++k) a1 += h0[k] * w1[k];
        h1[d] = tanhf(a1);
    }
    __syncthreads();
    if (d < 16) {
        float a2 = 0.f;
        const float* w2 = W2 + d * 32;
        for (int k = 0; k < 32; ++k) a2 += h1[k] * w2[k];
        logits[(size_t)rowid * 16 + d] = a2;
    }
}

// ---------------------------------------------------------------------------
// Softmax over the sequence axis (NP=2048) for one (b, head) per block.
// in/out strided by HEADS. grid = B*HEADS, block = 256. Aliasing-safe.
// ---------------------------------------------------------------------------
__global__ void softmax_seq_kernel(const float* __restrict__ in, float* __restrict__ out) {
    const int bh = blockIdx.x;
    const int b = bh >> 4;
    const int h = bh & 15;
    const float* src = in + (size_t)b * NP * HEADS + h;
    float* dst = out + (size_t)b * NP * HEADS + h;
    __shared__ float red[256];
    const int t = threadIdx.x;
    float mx = -1e30f;
    for (int s = t; s < NP; s += 256) mx = fmaxf(mx, src[(size_t)s * HEADS]);
    red[t] = mx;
    __syncthreads();
    for (int off = 128; off > 0; off >>= 1) {
        if (t < off) red[t] = fmaxf(red[t], red[t + off]);
        __syncthreads();
    }
    mx = red[0];
    __syncthreads();
    float sm = 0.f;
    for (int s = t; s < NP; s += 256) sm += __expf(src[(size_t)s * HEADS] - mx);
    red[t] = sm;
    __syncthreads();
    for (int off = 128; off > 0; off >>= 1) {
        if (t < off) red[t] += red[t + off];
        __syncthreads();
    }
    const float inv = 1.f / red[0];
    __syncthreads();
    for (int s = t; s < NP; s += 256)
        dst[(size_t)s * HEADS] = __expf(src[(size_t)s * HEADS] - mx) * inv;
}

// abar[b,s] = mean over heads of a[b,s,h]
__global__ void mean_heads_kernel(const float* __restrict__ a, float* __restrict__ abar) {
    int i = blockIdx.x * blockDim.x + threadIdx.x;
    if (i >= PROWS) return;
    float s = 0.f;
    for (int h = 0; h < HEADS; ++h) s += a[(size_t)i * HEADS + h];
    abar[i] = s * (1.f / HEADS);
}

// pooled[b,:] = sum_s abar[b,s] * dmat[b,s,:]; out = pooled @ mlp_W^T + mlp_b
__global__ void final_proj_kernel(const float* __restrict__ abar,
                                  const float* __restrict__ dmat,
                                  const float* __restrict__ mlp_W,
                                  const float* __restrict__ mlp_b,
                                  float* __restrict__ out) {
    const int b = blockIdx.x;
    const int d = threadIdx.x;    // 64 threads
    __shared__ float pooled[64];
    const float* ab = abar + (size_t)b * NP;
    const float* dm = dmat + (size_t)b * NP * 64;
    float acc = 0.f;
    for (int s = 0; s < NP; ++s) acc += ab[s] * dm[(size_t)s * 64 + d];
    pooled[d] = acc;
    __syncthreads();
    if (d < 32) {
        float o = mlp_b[d];
        const float* w = mlp_W + d * 64;
        for (int k = 0; k < 64; ++k) o += pooled[k] * w[k];
        out[b * 32 + d] = o;
    }
}

// ---------------------------------------------------------------------------
// Orchestration
// ---------------------------------------------------------------------------
extern "C" void kernel_launch(void* const* d_in, const int* in_sizes, int n_in,
                              void* d_out, int out_size, void* d_ws, size_t ws_size,
                              hipStream_t stream) {
    const float* pep_feat   = (const float*)d_in[0];
    const float* rec_feat   = (const float*)d_in[1];
    const float* edge_feat  = (const float*)d_in[2];
    const float* W_rec_esm  = (const float*)d_in[3];
    const float* W_pep_esm  = (const float*)d_in[4];
    const float* W_pep_emb  = (const float*)d_in[5];
    const float* W_rec_emb  = (const float*)d_in[6];
    const float* W_edge_emb = (const float*)d_in[7];
    const float* pep_ln_g   = (const float*)d_in[8];
    const float* pep_ln_b   = (const float*)d_in[9];
    const float* rec_ln_g   = (const float*)d_in[10];
    const float* rec_ln_b   = (const float*)d_in[11];
    const float* ga_Wp      = (const float*)d_in[12];
    const float* ga_bp      = (const float*)d_in[13];
    const float* ga_Wr      = (const float*)d_in[14];
    const float* ga_br      = (const float*)d_in[15];
    const float* ga_Wv      = (const float*)d_in[16];
    const float* ga_bv      = (const float*)d_in[17];
    const float* pool_ln_g  = (const float*)d_in[18];
    const float* pool_ln_b  = (const float*)d_in[19];
    const float* pool_W0    = (const float*)d_in[20];
    const float* pool_W1    = (const float*)d_in[21];
    const float* pool_W2    = (const float*)d_in[22];
    const float* mlp_W      = (const float*)d_in[23];
    const float* mlp_b      = (const float*)d_in[24];
    (void)in_sizes; (void)n_in; (void)out_size; (void)ws_size;

    char* ws = (char*)d_ws;
    size_t off = 0;
    auto alloc = [&](size_t bytes) -> void* {
        off = (off + 255) & ~(size_t)255;
        void* p = ws + off;
        off += bytes;
        return p;
    };

    float*  rec_esm  = (float*)alloc((size_t)RROWS * 64 * 4);
    float*  pep_esm  = (float*)alloc((size_t)PROWS * 64 * 4);
    __bf16* rec_edge = (__bf16*)alloc((size_t)RROWS * 128 * 2);
    float*  key      = (float*)alloc((size_t)RROWS * 64 * 4);
    float*  val      = (float*)alloc((size_t)RROWS * 64 * 4);
    float*  pep      = (float*)alloc((size_t)PROWS * 64 * 4);
    float*  q        = (float*)alloc((size_t)PROWS * 64 * 4);
    float*  dmat     = (float*)alloc((size_t)PROWS * 64 * 4);
    float*  logits   = (float*)alloc((size_t)PROWS * HEADS * 4);
    float*  abuf     = (float*)alloc((size_t)PROWS * HEADS * 4);
    float*  abar     = (float*)alloc((size_t)PROWS * 4);
    __bf16* fragRE   = (__bf16*)alloc((size_t)1280 * 64 * 2);
    __bf16* fragPE   = (__bf16*)alloc((size_t)1280 * 64 * 2);
    __bf16* fragR    = (__bf16*)alloc((size_t)128 * 64 * 2);
    __bf16* fragV    = (__bf16*)alloc((size_t)128 * 64 * 2);
    __bf16* fragP    = (__bf16*)alloc((size_t)64 * 64 * 2);

    // --- weight repack for the big ESM projections ---
    repack_w_kernel<<<(1280 * 64 + 255) / 256, 256, 0, stream>>>(W_rec_esm, fragRE, 1280);
    repack_w_kernel<<<(1280 * 64 + 255) / 256, 256, 0, stream>>>(W_pep_esm, fragPE, 1280);

    // --- big HBM-bound GEMMs via bf16 WMMA (on-the-fly f32->bf16) ---
    gemm_n64_kernel<float><<<RROWS / 128, 256, 0, stream>>>(
        rec_feat + 42, (long long)REC_F, fragRE, nullptr, rec_esm, 1280);
    gemm_n64_kernel<float><<<PROWS / 128, 256, 0, stream>>>(
        pep_feat + 27, (long long)PEP_F, fragPE, nullptr, pep_esm, 1280);

    // --- embeddings + LN + concat ---
    embed_rec_edge_kernel<<<RROWS, 64, 0, stream>>>(
        rec_feat, rec_esm, edge_feat, W_rec_emb, W_edge_emb, rec_ln_g, rec_ln_b, rec_edge);
    embed_pep_kernel<<<PROWS, 64, 0, stream>>>(
        pep_feat, pep_esm, W_pep_emb, pep_ln_g, pep_ln_b, pep);

    // --- GraphAttention layers ---
    for (int l = 0; l < NLAYERS; ++l) {
        repack_w_kernel<<<(128 * 64 + 255) / 256, 256, 0, stream>>>(ga_Wr + (size_t)l * 64 * 128, fragR, 128);
        repack_w_kernel<<<(128 * 64 + 255) / 256, 256, 0, stream>>>(ga_Wv + (size_t)l * 64 * 128, fragV, 128);
        repack_w_kernel<<<(64 * 64 + 255) / 256, 256, 0, stream>>>(ga_Wp + (size_t)l * 64 * 64, fragP, 64);

        gemm_n64_kernel<__bf16><<<RROWS / 128, 256, 0, stream>>>(
            rec_edge, 128LL, fragR, ga_br + (size_t)l * 64, key, 128);
        gemm_n64_kernel<__bf16><<<RROWS / 128, 256, 0, stream>>>(
            rec_edge, 128LL, fragV, ga_bv + (size_t)l * 64, val, 128);
        gemm_n64_kernel<float><<<PROWS / 128, 256, 0, stream>>>(
            pep, 64LL, fragP, ga_bp + (size_t)l * 64, q, 64);

        attention_kernel<<<PROWS, 64, 0, stream>>>(q, key, val, pep);
    }

    // --- pooling head ---
    pool_mlp_kernel<<<PROWS, 64, 0, stream>>>(
        pep, pool_ln_g, pool_ln_b, pool_W0, pool_W1, pool_W2, dmat, logits);
    softmax_seq_kernel<<<BB * HEADS, 256, 0, stream>>>(logits, abuf);
    softmax_seq_kernel<<<BB * HEADS, 256, 0, stream>>>(abuf, abuf);
    mean_heads_kernel<<<PROWS / 256, 256, 0, stream>>>(abuf, abar);
    final_proj_kernel<<<BB, 64, 0, stream>>>(abar, dmat, mlp_W, mlp_b, (float*)d_out);
}
